// BasicBlock_1w1a_77034533421486
// MI455X (gfx1250) — 8-bit matrix operand order repaired and checked by reading, compiled, not run on hardware
//
#include <hip/hip_runtime.h>
#include <hip/hip_bf16.h>

typedef int v8i __attribute__((ext_vector_type(8)));
typedef __attribute__((__vector_size__(4 * sizeof(int)))) int v4i_b;

#define N_IMG   64
#define C_CH    256
#define HW      28
#define PLANE   (HW * HW)
#define M_TOTAL (N_IMG * PLANE)
#define K_TOTAL (C_CH * 9)
#define ELEMS   ((long)N_IMG * C_CH * PLANE)
#define CNT_F   ((double)(N_IMG * PLANE))
#define BN_EPS  1e-5f

#define B_ROW_PAD 80

#if __has_builtin(__builtin_amdgcn_global_load_async_to_lds_b128)
#define HAVE_ASYNC_LDS 1
#else
#define HAVE_ASYNC_LDS 0
#endif

__device__ __forceinline__ void wait_async_all() {
#if __has_builtin(__builtin_amdgcn_s_wait_asynccnt)
    __builtin_amdgcn_s_wait_asynccnt(0);
#else
    asm volatile("s_wait_asynccnt 0x0" ::: "memory");
#endif
}

__global__ void k_zero_stats(unsigned long long* p) {
    int i = blockIdx.x * blockDim.x + threadIdx.x;
    if (i < 1024) p[i] = 0ull;
}

__global__ __launch_bounds__(256) void k_binarize_input(const float* __restrict__ x,
                                                        signed char* __restrict__ xb) {
    long i = (long)blockIdx.x * 256 + threadIdx.x;
    if (i >= ELEMS) return;
    int n = (int)(i / (C_CH * PLANE));
    int r = (int)(i - (long)n * (C_CH * PLANE));
    int c = r / PLANE;
    int p = r - c * PLANE;
    float v = x[i];
    xb[((long)n * PLANE + p) * C_CH + c] = (v > 0.f) ? 1 : ((v < 0.f) ? -1 : 0);
}

__global__ __launch_bounds__(256) void k_binarize_weights(const float* __restrict__ w,
                                                          signed char* __restrict__ wb) {
    int i = blockIdx.x * 256 + threadIdx.x;
    int co  = i / K_TOTAL;
    int r   = i - co * K_TOTAL;
    int ci  = r / 9;
    int tap = r - ci * 9;
    float v = w[i];
    wb[(long)co * K_TOTAL + tap * C_CH + ci] = (v > 0.f) ? 1 : ((v < 0.f) ? -1 : 0);
}

__global__ __launch_bounds__(256) void k_bconv_wmma(const signed char* __restrict__ xb,
                                                    const signed char* __restrict__ wb,
                                                    short* __restrict__ y) {
    __shared__ signed char Bs[2][64 * B_ROW_PAD];

    const int tid  = threadIdx.x;
    const int lane = tid & 31;
    const int wave = tid >> 5;
    const int m0    = blockIdx.x * 128 + wave * 16;
    const int nbase = blockIdx.y * 64;

    const int srow = tid >> 2;
    const int sseg = (tid & 3) << 4;
    const signed char* sBase = wb + (long)(nbase + srow) * K_TOTAL + sseg;

    const int row = m0 + (lane & 15);
    const int hiA = (lane >> 4) << 4;
    const int n   = row / PLANE;
    const int rem = row - n * PLANE;
    const int h   = rem / HW;
    const int w   = rem - h * HW;

    const int colLocal = lane & 15;
    const int hiB = (lane >> 4) << 4;

    v8i acc[4];
    #pragma unroll
    for (int t = 0; t < 4; ++t) acc[t] = v8i{0,0,0,0,0,0,0,0};

    auto stage = [&](int kt, int buf) {
        const signed char* gp = sBase + kt * 64;
        signed char* lp = &Bs[buf][srow * B_ROW_PAD + sseg];
#if HAVE_ASYNC_LDS
        __builtin_amdgcn_global_load_async_to_lds_b128(
            (v4i_b*)gp, (v4i_b*)lp, 0, 0);
#else
        *(int4*)lp = *(const int4*)gp;
#endif
    };

    stage(0, 0);
    wait_async_all();
    __syncthreads();

    for (int kt = 0; kt < 36; ++kt) {
        const int cur = kt & 1;
        if (kt + 1 < 36) stage(kt + 1, cur ^ 1);

        const int tap = kt >> 2;
        const int cb  = (kt & 3) << 6;
        const int dy  = tap / 3;
        const int dx  = tap - dy * 3;
        const int ih  = h + dy - 1;
        const int iw  = w + dx - 1;
        const bool inb = ((unsigned)ih < (unsigned)HW) & ((unsigned)iw < (unsigned)HW);

        v8i a;
        const signed char* ap =
            xb + ((((long)n * HW + ih) * HW + iw) * C_CH + cb + hiA);
        #pragma unroll
        for (int j = 0; j < 4; ++j) {
            unsigned long long v =
                inb ? *(const unsigned long long*)(ap + (j >> 1) * 32 + (j & 1) * 8) : 0ull;
            a[2 * j]     = (int)(unsigned)(v & 0xffffffffull);
            a[2 * j + 1] = (int)(unsigned)(v >> 32);
        }

        #pragma unroll
        for (int nt = 0; nt < 4; ++nt) {
            const int clocal = nt * 16 + colLocal;
            const signed char* bp = &Bs[cur][clocal * B_ROW_PAD + hiB];
            const int4 q0 = *(const int4*)bp;
            const int4 q1 = *(const int4*)(bp + 32);
            v8i b;
            b[0] = q0.x; b[1] = q0.y; b[2] = q0.z; b[3] = q0.w;
            b[4] = q1.x; b[5] = q1.y; b[6] = q1.z; b[7] = q1.w;
            acc[nt] = __builtin_amdgcn_wmma_i32_16x16x64_iu8(
                true, a, true, b, acc[nt], false, false);
        }

        wait_async_all();
        __syncthreads();
    }

    const int rBase = m0 + ((lane >> 4) << 3);
    #pragma unroll
    for (int nt = 0; nt < 4; ++nt) {
        const int c = nbase + nt * 16 + colLocal;
        #pragma unroll
        for (int v = 0; v < 8; ++v) {
            y[(long)(rBase + v) * C_CH + c] = (short)acc[nt][v];
        }
    }
}

__global__ __launch_bounds__(256) void k_bn_stats(const short* __restrict__ y,
                                                  unsigned long long* __restrict__ S,
                                                  unsigned long long* __restrict__ SS) {
    const long stride = (long)gridDim.x * 256;
    long i = (long)blockIdx.x * 256 + threadIdx.x;
    const int c = (int)(i & 255);
    long long ls = 0, lq = 0;
    for (; i < ELEMS; i += stride) {
        int v = y[i];
        ls += v;
        lq += (long long)v * v;
    }
    atomicAdd(&S[c],  (unsigned long long)ls);
    atomicAdd(&SS[c], (unsigned long long)lq);
}

__global__ void k_bn_params(const unsigned long long* __restrict__ S,
                            const unsigned long long* __restrict__ SS,
                            const float* __restrict__ gamma,
                            const float* __restrict__ beta,
                            float* __restrict__ scale,
                            float* __restrict__ shift) {
    int c = threadIdx.x;
    if (c >= C_CH) return;
    double mean = (double)(long long)S[c] / CNT_F;
    double var  = (double)(long long)SS[c] / CNT_F - mean * mean;
    float sc = gamma[c] * rsqrtf((float)var + BN_EPS);
    scale[c] = sc;
    shift[c] = beta[c] - (float)mean * sc;
}

__global__ __launch_bounds__(256) void k_binarize_bn(const short* __restrict__ y,
                                                     const float* __restrict__ scale,
                                                     const float* __restrict__ shift,
                                                     signed char* __restrict__ xb) {
    long i = (long)blockIdx.x * 256 + threadIdx.x;
    if (i >= ELEMS) return;
    int c = (int)(i & 255);
    float v = (float)y[i] * scale[c] + shift[c];
    xb[i] = (v > 0.f) ? 1 : ((v < 0.f) ? -1 : 0);
}

__global__ __launch_bounds__(256) void k_finalize(const short* __restrict__ y2,
                                                  const float* __restrict__ scale,
                                                  const float* __restrict__ shift,
                                                  const float* __restrict__ x,
                                                  float* __restrict__ out) {
    long i = (long)blockIdx.x * 256 + threadIdx.x;
    if (i >= ELEMS) return;
    int n = (int)(i / (C_CH * PLANE));
    int r = (int)(i - (long)n * (C_CH * PLANE));
    int c = r / PLANE;
    int p = r - c * PLANE;
    long nhwc = ((long)n * PLANE + p) * C_CH + c;
    out[i] = (float)y2[nhwc] * scale[c] + shift[c] + x[i];
}

extern "C" void kernel_launch(void* const* d_in, const int* in_sizes, int n_in,
                              void* d_out, int out_size, void* d_ws, size_t ws_size,
                              hipStream_t stream) {
    (void)in_sizes; (void)n_in; (void)out_size;
    const float* x  = (const float*)d_in[0];
    const float* w1 = (const float*)d_in[1];
    const float* g1 = (const float*)d_in[2];
    const float* b1 = (const float*)d_in[3];
    const float* w2 = (const float*)d_in[4];
    const float* g2 = (const float*)d_in[5];
    const float* b2 = (const float*)d_in[6];
    float* out = (float*)d_out;

    char* ws = (char*)d_ws;
    size_t off = 0;
    auto take = [&](size_t bytes) -> char* {
        char* p = ws + off;
        off = (off + bytes + 255) & ~(size_t)255;
        return p;
    };
    signed char* wb1 = (signed char*)take(256 * (size_t)K_TOTAL);
    signed char* wb2 = (signed char*)take(256 * (size_t)K_TOTAL);
    unsigned long long* stats = (unsigned long long*)take(4 * 256 * 8);
    unsigned long long* S1  = stats;
    unsigned long long* SS1 = stats + 256;
    unsigned long long* S2  = stats + 512;
    unsigned long long* SS2 = stats + 768;
    float* scale1 = (float*)take(256 * 4);
    float* shift1 = (float*)take(256 * 4);
    float* scale2 = (float*)take(256 * 4);
    float* shift2 = (float*)take(256 * 4);
    signed char* xb1 = (signed char*)take((size_t)ELEMS);
    signed char* xb2 = (signed char*)take((size_t)ELEMS);
    short* y1 = (short*)take((size_t)ELEMS * 2);
    short* y2 = (short*)take((size_t)ELEMS * 2);
    (void)ws_size;

    const int eltBlocks = (int)((ELEMS + 255) / 256);
    const dim3 convGrid(M_TOTAL / 128, C_CH / 64);

    k_zero_stats<<<4, 256, 0, stream>>>(stats);

    k_binarize_input<<<eltBlocks, 256, 0, stream>>>(x, xb1);
    k_binarize_weights<<<K_TOTAL, 256, 0, stream>>>(w1, wb1);
    k_binarize_weights<<<K_TOTAL, 256, 0, stream>>>(w2, wb2);

    k_bconv_wmma<<<convGrid, 256, 0, stream>>>(xb1, wb1, y1);
    k_bn_stats<<<512, 256, 0, stream>>>(y1, S1, SS1);
    k_bn_params<<<1, 256, 0, stream>>>(S1, SS1, g1, b1, scale1, shift1);
    k_binarize_bn<<<eltBlocks, 256, 0, stream>>>(y1, scale1, shift1, xb2);

    k_bconv_wmma<<<convGrid, 256, 0, stream>>>(xb2, wb2, y2);
    k_bn_stats<<<512, 256, 0, stream>>>(y2, S2, SS2);
    k_bn_params<<<1, 256, 0, stream>>>(S2, SS2, g2, b2, scale2, shift2);

    k_finalize<<<eltBlocks, 256, 0, stream>>>(y2, scale2, shift2, x, out);
}
